// RGCNModel_21921513079385
// MI455X (gfx1250) — compile-verified, matrix operand order checked by reading, not run on hardware
//
#include <hip/hip_runtime.h>
#include <hip/hip_bf16.h>

// ---- problem constants (from reference) ----
#define NN    60000
#define NPAD  60032      // multiple of 128 -> guard-free WMMA GEMM blocks of 128 rows
#define EE    600000
#define RR    7
#define DIN   64
#define DH    128
#define DC    64

typedef __attribute__((ext_vector_type(16))) __bf16 bf16x16;
typedef __attribute__((ext_vector_type(8)))  float  f32x8;

__device__ __forceinline__ unsigned short f2bf(float f) {
    unsigned int u = __float_as_uint(f);
    unsigned int r = u + 0x7FFFu + ((u >> 16) & 1u);   // round-to-nearest-even
    return (unsigned short)(r >> 16);
}
__device__ __forceinline__ float bf2f(unsigned short s) {
    return __uint_as_float(((unsigned int)s) << 16);
}

// ---------------- weight pre-pack into WMMA B-operand lane layout ----------------
// dword index ((kt*NT + nt)*32 + lane)*8 + v ; each dword = bf16{W[k,n], W[k+1,n]}
// matching the 16-bit 32x16 B VGPR layout of v_wmma_f32_16x16x32_bf16.
__global__ void pack_w_kernel(const float* __restrict__ W, unsigned int* __restrict__ P,
                              int K, int NCOLS) {
    int NT = NCOLS >> 4;
    int KT = K >> 5;
    int total = KT * NT * 256;
    int idx = blockIdx.x * blockDim.x + threadIdx.x;
    if (idx >= total) return;
    int v    = idx & 7;
    int lane = (idx >> 3) & 31;
    int nt   = (idx >> 8) % NT;
    int kt   = idx / (256 * NT);
    int g = v >> 2, vv = v & 3, hf = lane >> 4;
    int k = kt * 32 + g * 16 + hf * 8 + vv * 2;
    int n = nt * 16 + (lane & 15);
    unsigned short lo = f2bf(W[(size_t)k * NCOLS + n]);
    unsigned short hi = f2bf(W[(size_t)(k + 1) * NCOLS + n]);
    P[idx] = (unsigned int)lo | ((unsigned int)hi << 16);
}

// ---------------- clamp(x,-10,10) -> bf16, zero-fill pad rows ----------------
__global__ void clip_bf16_kernel(const float* __restrict__ x, unsigned short* __restrict__ xb) {
    int i = blockIdx.x * blockDim.x + threadIdx.x;
    if (i >= NPAD * DIN) return;
    int r = i / DIN;
    float f = 0.0f;
    if (r < NN) f = fminf(10.0f, fmaxf(-10.0f, x[i]));
    xb[i] = f2bf(f);
}

// ---------------- WMMA bf16 GEMM: C[NPAD x NT*16] = A[NPAD x K] * Bpacked ----------------
// Block = 4 waves, 128 rows. A tile (128 x K bf16) async-staged into LDS (padded rows),
// each wave owns a 32/16-col strip with B fragments register-resident across all of K.
// HAS_BIAS / WF (f32 out) / WB (bf16 out) are compile-time -> branch-free epilogue.
template <int NT_TOTAL, int K, bool HAS_BIAS, bool WF, bool WB>
__global__ __launch_bounds__(128)
void gemm_bf16_kernel(const unsigned short* __restrict__ A,     // bf16 row-major [NPAD x K]
                      const uint4* __restrict__ Bp,             // packed weights
                      const float* __restrict__ bias,           // [NT_TOTAL*16]
                      float* __restrict__ Cf,                   // f32 out
                      unsigned short* __restrict__ Cb)          // bf16 out
{
    constexpr int NW     = NT_TOTAL / 4;   // n-tiles per wave
    constexpr int KT     = K / 32;         // k-steps
    constexpr int STRIDE = K + 8;          // +16B row pad -> conflict-free half-wave reads
    constexpr int W      = NT_TOTAL * 16;  // C row width

    __shared__ __align__(16) unsigned short ldsA[128 * STRIDE];

    const int tid  = threadIdx.x;
    const int wave = tid >> 5;
    const int lane = tid & 31;
    const int hf   = lane >> 4;
    const int row0 = blockIdx.x * 128;

    // ---- stage A tile into LDS via CDNA5 async copy (ASYNCcnt-tracked) ----
    {
        constexpr int CPR   = K / 8;          // b128 chunks per row
        constexpr int TOTAL = 128 * CPR;      // multiple of 128 -> uniform loop
        const unsigned ldsbase = (unsigned)(uintptr_t)(&ldsA[0]);
#pragma unroll
        for (int c = tid; c < TOTAL; c += 128) {
            int r  = c / CPR;
            int kc = c - r * CPR;
            unsigned long long ga =
                (unsigned long long)(uintptr_t)(A + (size_t)(row0 + r) * K + kc * 8);
            unsigned la = ldsbase + (unsigned)((r * STRIDE + kc * 8) * 2);
            asm volatile("global_load_async_to_lds_b128 %0, %1, off"
                         :: "v"(la), "v"(ga) : "memory");
        }
        asm volatile("s_wait_asynccnt 0" ::: "memory");
        __syncthreads();
    }

    // ---- preload this wave's B fragments (register-resident for the whole kernel) ----
    union Frag { uint4 u[2]; bf16x16 v; };
    Frag b[KT][NW];
#pragma unroll
    for (int kt = 0; kt < KT; ++kt)
#pragma unroll
        for (int j = 0; j < NW; ++j) {
            const uint4* p = Bp + ((size_t)(kt * NT_TOTAL + wave * NW + j) * 32 + lane) * 2;
            b[kt][j].u[0] = p[0];
            b[kt][j].u[1] = p[1];
        }

    // ---- hoist bias (row-invariant) ----
    float bv[NW];
#pragma unroll
    for (int j = 0; j < NW; ++j)
        bv[j] = HAS_BIAS ? bias[(wave * NW + j) * 16 + (lane & 15)] : 0.0f;

    // ---- 8 m-tiles of 16 rows, A from LDS, B from registers ----
#pragma unroll 1
    for (int mt = 0; mt < 8; ++mt) {
        f32x8 acc[NW];
        f32x8 zero = {0.f, 0.f, 0.f, 0.f, 0.f, 0.f, 0.f, 0.f};
#pragma unroll
        for (int j = 0; j < NW; ++j) acc[j] = zero;

        const unsigned short* arow = ldsA + (mt * 16 + (lane & 15)) * STRIDE;
#pragma unroll
        for (int kt = 0; kt < KT; ++kt) {
            int kk = kt * 32;
            Frag a;
            a.u[0] = *(const uint4*)(arow + kk + hf * 8);
            a.u[1] = *(const uint4*)(arow + kk + 16 + hf * 8);
#pragma unroll
            for (int j = 0; j < NW; ++j)
                acc[j] = __builtin_amdgcn_wmma_f32_16x16x32_bf16(
                    false, a.v, false, b[kt][j].v, (short)0, acc[j], false, false);
        }

        // epilogue: D layout -> vgpr i = row (mt*16 + hf*8 + i), col (lane&15) of tile
        const int rbase = row0 + mt * 16 + hf * 8;
#pragma unroll
        for (int j = 0; j < NW; ++j) {
            const int n = (wave * NW + j) * 16 + (lane & 15);
            if (WF) {
                float* cp = Cf + (size_t)rbase * W + n;
#pragma unroll
                for (int i = 0; i < 8; i++) cp[(size_t)i * W] = acc[j][i] + bv[j];
            }
            if (WB) {
                unsigned short* cp = Cb + (size_t)rbase * W + n;
#pragma unroll
                for (int i = 0; i < 8; i++) cp[(size_t)i * W] = f2bf(acc[j][i] + bv[j]);
            }
        }
    }
}

// ---------------- edge-count (for per-(rel,dst) mean) ----------------
__global__ void zero_kernel(float* __restrict__ p, int n) {
    int i = blockIdx.x * blockDim.x + threadIdx.x;
    if (i < n) p[i] = 0.0f;
}
__global__ void count_kernel(const int* __restrict__ et, const int* __restrict__ ei,
                             float* __restrict__ cnt) {
    int e = blockIdx.x * blockDim.x + threadIdx.x;
    if (e >= EE) return;
    atomicAdd(&cnt[(size_t)et[e] * NN + ei[EE + e]], 1.0f);
}
__global__ void invert_kernel(float* __restrict__ c, int n) {
    int i = blockIdx.x * blockDim.x + threadIdx.x;
    if (i < n) c[i] = 1.0f / fmaxf(c[i], 1.0f);
}

// ---------------- scatter: out[dst] += inv[r,dst] * t_r[src] ----------------
// 16 threads per edge, 8 bf16 elements (one b128 gather) per thread.
__global__ __launch_bounds__(256)
void scatter_kernel(const int* __restrict__ ei, const int* __restrict__ et,
                    const unsigned short* __restrict__ T, const float* __restrict__ inv,
                    float* __restrict__ out, int r) {
    int gid = blockIdx.x * blockDim.x + threadIdx.x;
    int e   = gid >> 4;
    int seg = gid & 15;
    if (e >= EE) return;
    if (et[e] != r) return;
    int s = ei[e];
    int d = ei[EE + e];
    float w = inv[(size_t)r * NN + d];
    uint4 tv = *(const uint4*)(T + (size_t)s * DH + seg * 8);
    float* o = out + (size_t)d * DH + seg * 8;
    unsigned int u[4] = {tv.x, tv.y, tv.z, tv.w};
#pragma unroll
    for (int j = 0; j < 4; j++) {
        atomicAdd(o + 2 * j,     w * bf2f((unsigned short)(u[j] & 0xFFFFu)));
        atomicAdd(o + 2 * j + 1, w * bf2f((unsigned short)(u[j] >> 16)));
    }
}

// ---------------- LayerNorm + ReLU (D=128), wave per row, write bf16 h ----------------
__global__ __launch_bounds__(256)
void ln_relu128_kernel(const float* __restrict__ in, const float* __restrict__ g,
                       const float* __restrict__ b, unsigned short* __restrict__ hb) {
    int lane = threadIdx.x & 31;
    int row  = blockIdx.x * 8 + (threadIdx.x >> 5);
    if (row >= NPAD) return;
    int n = lane * 4;
    if (row >= NN) {  // keep pad rows finite-zero for the next GEMM's A operand
        *(uint2*)(hb + (size_t)row * DH + n) = make_uint2(0u, 0u);
        return;
    }
    float4 x = *(const float4*)(in + (size_t)row * DH + n);
    float s = x.x + x.y + x.z + x.w;
#pragma unroll
    for (int o = 16; o > 0; o >>= 1) s += __shfl_xor(s, o, 32);
    float mean = s * (1.0f / 128.0f);
    float d0 = x.x - mean, d1 = x.y - mean, d2 = x.z - mean, d3 = x.w - mean;
    float ss = d0 * d0 + d1 * d1 + d2 * d2 + d3 * d3;
#pragma unroll
    for (int o = 16; o > 0; o >>= 1) ss += __shfl_xor(ss, o, 32);
    float rstd = rsqrtf(ss * (1.0f / 128.0f) + 1e-5f);
    float y0 = fmaxf(d0 * rstd * g[n + 0] + b[n + 0], 0.0f);
    float y1 = fmaxf(d1 * rstd * g[n + 1] + b[n + 1], 0.0f);
    float y2 = fmaxf(d2 * rstd * g[n + 2] + b[n + 2], 0.0f);
    float y3 = fmaxf(d3 * rstd * g[n + 3] + b[n + 3], 0.0f);
    unsigned int lo = (unsigned int)f2bf(y0) | ((unsigned int)f2bf(y1) << 16);
    unsigned int hi = (unsigned int)f2bf(y2) | ((unsigned int)f2bf(y3) << 16);
    *(uint2*)(hb + (size_t)row * DH + n) = make_uint2(lo, hi);
}

// ---------------- LayerNorm + ReLU (D=64) in-place f32 (classifier hidden) ----------------
__global__ __launch_bounds__(256)
void ln_relu64_kernel(const float* __restrict__ in, const float* __restrict__ g,
                      const float* __restrict__ b, float* __restrict__ z) {
    int lane = threadIdx.x & 31;
    int row  = blockIdx.x * 8 + (threadIdx.x >> 5);
    if (row >= NN) return;
    int n = lane * 2;
    float2 x = *(const float2*)(in + (size_t)row * DC + n);
    float s = x.x + x.y;
#pragma unroll
    for (int o = 16; o > 0; o >>= 1) s += __shfl_xor(s, o, 32);
    float mean = s * (1.0f / 64.0f);
    float d0 = x.x - mean, d1 = x.y - mean;
    float ss = d0 * d0 + d1 * d1;
#pragma unroll
    for (int o = 16; o > 0; o >>= 1) ss += __shfl_xor(ss, o, 32);
    float rstd = rsqrtf(ss * (1.0f / 64.0f) + 1e-5f);
    float2 y;
    y.x = fmaxf(d0 * rstd * g[n + 0] + b[n + 0], 0.0f);
    y.y = fmaxf(d1 * rstd * g[n + 1] + b[n + 1], 0.0f);
    *(float2*)(z + (size_t)row * DC + n) = y;
}

// ---------------- final matvec [N,64] @ [64,1] + b ----------------
__global__ void final_kernel(const float* __restrict__ z, const float* __restrict__ W2,
                             const float* __restrict__ b2, float* __restrict__ out) {
    int nid = blockIdx.x * blockDim.x + threadIdx.x;
    if (nid >= NN) return;
    float acc = b2[0];
    const float* zr = z + (size_t)nid * DC;
#pragma unroll
    for (int d = 0; d < DC; d++) acc += zr[d] * W2[d];
    out[nid] = acc;
}

// ---------------- host orchestration ----------------
extern "C" void kernel_launch(void* const* d_in, const int* in_sizes, int n_in,
                              void* d_out, int out_size, void* d_ws, size_t ws_size,
                              hipStream_t stream) {
    (void)in_sizes; (void)n_in; (void)out_size; (void)ws_size;
    const float* x       = (const float*)d_in[0];
    const int*   ei      = (const int*)d_in[1];      // [2,E]
    const int*   et      = (const int*)d_in[2];      // [E]
    const float* emb_W   = (const float*)d_in[3];
    const float* emb_b   = (const float*)d_in[4];
    const float* rel_W   = (const float*)d_in[5];    // [L,R,128,128]
    const float* root_W  = (const float*)d_in[6];    // [L,128,128]
    const float* conv_b  = (const float*)d_in[7];    // [L,128]
    const float* ln_g    = (const float*)d_in[8];
    const float* ln_b    = (const float*)d_in[9];
    const float* cls_W1  = (const float*)d_in[10];
    const float* cls_b1  = (const float*)d_in[11];
    const float* cls_lng = (const float*)d_in[12];
    const float* cls_lnb = (const float*)d_in[13];
    const float* cls_W2  = (const float*)d_in[14];
    const float* cls_b2  = (const float*)d_in[15];
    float* out = (float*)d_out;

    // workspace carve (~90 MB total)
    char* ws = (char*)d_ws;
    size_t off = 0;
    auto carve = [&](size_t bytes) -> void* {
        void* p = ws + off;
        off += (bytes + 255) & ~(size_t)255;
        return p;
    };
    unsigned short* x_bf  = (unsigned short*)carve((size_t)NPAD * DIN * 2);
    unsigned short* h_bf  = (unsigned short*)carve((size_t)NPAD * DH * 2);
    unsigned short* t_bf  = (unsigned short*)carve((size_t)NPAD * DH * 2);
    float*          conv  = (float*)carve((size_t)NPAD * DH * 4);
    float*          zbuf  = (float*)carve((size_t)NPAD * DC * 4);
    float*          cnt   = (float*)carve((size_t)RR * NN * 4);
    unsigned int*   p_emb = (unsigned int*)carve((size_t)2 * 8 * 256 * 4);    // K=64,N=128
    unsigned int*   p_rel = (unsigned int*)carve((size_t)2 * RR * 8192 * 4);  // 14 x (128,128)
    unsigned int*   p_root= (unsigned int*)carve((size_t)2 * 8192 * 4);       // 2 x (128,128)
    unsigned int*   p_cls = (unsigned int*)carve((size_t)4096 * 4);           // K=128,N=64

    // 1) pack all weights into WMMA B lane layout
    pack_w_kernel<<<(4096 + 255) / 256, 256, 0, stream>>>(emb_W, p_emb, DIN, DH);
    for (int l = 0; l < 2; l++) {
        for (int r = 0; r < RR; r++)
            pack_w_kernel<<<(8192 + 255) / 256, 256, 0, stream>>>(
                rel_W + (size_t)(l * RR + r) * DH * DH, p_rel + (size_t)(l * RR + r) * 8192,
                DH, DH);
        pack_w_kernel<<<(8192 + 255) / 256, 256, 0, stream>>>(
            root_W + (size_t)l * DH * DH, p_root + (size_t)l * 8192, DH, DH);
    }
    pack_w_kernel<<<(4096 + 255) / 256, 256, 0, stream>>>(cls_W1, p_cls, DH, DC);

    // 2) node embedding: h = clip(x) @ emb_W + emb_b  (bf16 activations)
    clip_bf16_kernel<<<(NPAD * DIN + 255) / 256, 256, 0, stream>>>(x, x_bf);
    gemm_bf16_kernel<8, DIN, true, false, true><<<NPAD / 128, 128, 0, stream>>>(
        x_bf, (const uint4*)p_emb, emb_b, nullptr, h_bf);

    // 3) per-(relation,dst) inverse neighbor counts (shared by both layers)
    zero_kernel<<<(RR * NN + 255) / 256, 256, 0, stream>>>(cnt, RR * NN);
    count_kernel<<<(EE + 255) / 256, 256, 0, stream>>>(et, ei, cnt);
    invert_kernel<<<(RR * NN + 255) / 256, 256, 0, stream>>>(cnt, RR * NN);

    // 4) RGCN layers: conv = h@Wroot + b + sum_r scatter(inv * (h@Wrel[r])[src] -> dst)
    for (int l = 0; l < 2; l++) {
        gemm_bf16_kernel<8, DH, true, true, false><<<NPAD / 128, 128, 0, stream>>>(
            h_bf, (const uint4*)(p_root + (size_t)l * 8192), conv_b + (size_t)l * DH,
            conv, nullptr);
        for (int r = 0; r < RR; r++) {
            gemm_bf16_kernel<8, DH, false, false, true><<<NPAD / 128, 128, 0, stream>>>(
                h_bf, (const uint4*)(p_rel + (size_t)(l * RR + r) * 8192), nullptr,
                nullptr, t_bf);
            scatter_kernel<<<(EE * 16 + 255) / 256, 256, 0, stream>>>(
                ei, et, t_bf, cnt, conv, r);
        }
        ln_relu128_kernel<<<NPAD / 8, 256, 0, stream>>>(
            conv, ln_g + (size_t)l * DH, ln_b + (size_t)l * DH, h_bf);
    }

    // 5) classifier: relu(LN(h@W1+b1)) @ W2 + b2
    gemm_bf16_kernel<4, DH, true, true, false><<<NPAD / 128, 128, 0, stream>>>(
        h_bf, (const uint4*)p_cls, cls_b1, zbuf, nullptr);
    ln_relu64_kernel<<<(NN + 7) / 8, 256, 0, stream>>>(zbuf, cls_lng, cls_lnb, zbuf);
    final_kernel<<<(NN + 255) / 256, 256, 0, stream>>>(zbuf, cls_W2, cls_b2, out);
}
